// GATModel_39986145525987
// MI455X (gfx1250) — matrix operand order repaired and checked by reading, not compiled, not run on hardware
//
#include <hip/hip_runtime.h>

// ---------------------------------------------------------------------------
// Types for CDNA5 WMMA (wave32, 16x16x32 bf16 -> f32)
// ---------------------------------------------------------------------------
typedef __attribute__((ext_vector_type(16))) __bf16 v16bf;
typedef __attribute__((ext_vector_type(8)))  __bf16 v8bf;
typedef __attribute__((ext_vector_type(8)))  float  v8f;

union ABFrag { v16bf v; v8bf h[2]; };

__device__ __forceinline__ __bf16 f2bf(float f) {
    union { float f; unsigned u; } in; in.f = f;
    unsigned r = (in.u + 0x7FFFu + ((in.u >> 16) & 1u)) >> 16;   // RNE
    union { unsigned short s; __bf16 b; } out; out.s = (unsigned short)r;
    return out.b;
}

// Monotonic float <-> uint mapping for atomic max on signed floats
__device__ __forceinline__ unsigned f2ord(float f) {
    unsigned u = __float_as_uint(f);
    return (u & 0x80000000u) ? ~u : (u | 0x80000000u);
}
__device__ __forceinline__ float ord2f(unsigned u) {
    unsigned b = (u & 0x80000000u) ? (u & 0x7FFFFFFFu) : ~u;
    return __uint_as_float(b);
}

#define NEG_SLOPE 0.2f

// ---------------------------------------------------------------------------
// GEMM: feat[N,128] = X[N,128] @ W[128,128] via v_wmma_f32_16x16x32_bf16.
// Block = 128 threads (4 waves). Block tile = 64 rows x 128 cols.
// Wave w handles rows 16w..16w+15 across eight 16-col tiles; K=128 in 4 steps.
// LDS: X rows (row-major) and W transposed (col-of-W contiguous), stride 136
// bf16 (272B = 68 dwords) -> 16 lanes * 16B fragment loads hit all 64 banks.
// ---------------------------------------------------------------------------
__global__ __launch_bounds__(128)
void gat_gemm_wmma(const float* __restrict__ X, const float* __restrict__ Wm,
                   float* __restrict__ out, int Nrows) {
    __shared__ __bf16 sA[64 * 136];    // X tile  [r][k]
    __shared__ __bf16 sB[128 * 136];   // W^T     [n][k]

    const int tid  = threadIdx.x;       // 0..127
    const int lane = tid & 31;
    const int wave = tid >> 5;          // 0..3
    const int rowBase = blockIdx.x * 64;

    // Stage W transposed: sB[n][k] = bf16(W[k*128+n]) (coalesced global reads)
    for (int i = tid; i < 128 * 128; i += 128) {
        int k = i >> 7, n = i & 127;
        sB[n * 136 + k] = f2bf(Wm[i]);
    }
    // Stage X rows (zero-pad past N)
    for (int i = tid; i < 64 * 128; i += 128) {
        int r = i >> 7, k = i & 127;
        int gr = rowBase + r;
        float v = (gr < Nrows) ? X[(size_t)gr * 128 + k] : 0.0f;
        sA[r * 136 + k] = f2bf(v);
    }
    __syncthreads();

    v8f acc[8] = {};

    // ISA 7.12.2 bf16 A layout: lanes 0-15 row M=lane, halves 0..7 -> K 0..7,
    // halves 8..15 -> K 16..23; lanes 16-31 row M=lane-16, K 8..15 / 24..31.
    const int rA   = (wave << 4) + (lane & 15);
    const int kloA = (lane < 16) ? 0 : 8;
    // B layout (32x16): lanes 0-15 col N=lane hold K 0..7 / 16..23; lanes 16-31 hold K 8..15 / 24..31.
    const int kloB = (lane < 16) ? 0 : 8;
    const int nB   = lane & 15;

#pragma unroll
    for (int kt = 0; kt < 4; ++kt) {
        ABFrag a;
        const int abase = rA * 136 + kt * 32 + kloA;
        a.h[0] = *(const v8bf*)&sA[abase];        // K +0..7
        a.h[1] = *(const v8bf*)&sA[abase + 16];   // K +16..23
#pragma unroll
        for (int c = 0; c < 8; ++c) {
            ABFrag b;
            const int bbase = (c * 16 + nB) * 136 + kt * 32 + kloB;
            b.h[0] = *(const v8bf*)&sB[bbase];
            b.h[1] = *(const v8bf*)&sB[bbase + 16];
            acc[c] = __builtin_amdgcn_wmma_f32_16x16x32_bf16(
                false, a.v, false, b.v, (short)0, acc[c], false, false);
        }
    }

    // C/D layout: VGPR r -> M=r (lanes 0-15) / M=r+8 (lanes 16-31), N=lane&15
    const int rOut = rowBase + (wave << 4) + ((lane >> 4) << 3);
    const int nOut = lane & 15;
#pragma unroll
    for (int c = 0; c < 8; ++c) {
#pragma unroll
        for (int r = 0; r < 8; ++r) {
            int gr = rOut + r;
            if (gr < Nrows) out[(size_t)gr * 128 + c * 16 + nOut] = acc[c][r];
        }
    }
}

// ---------------------------------------------------------------------------
// el/er: per (node, head) dot of feat[n,h,:] with attn vectors (D=16)
// ---------------------------------------------------------------------------
__global__ void gat_attn(const float* __restrict__ feat,
                         const float* __restrict__ al, const float* __restrict__ ar,
                         float* __restrict__ el, float* __restrict__ er, int N) {
    int i = blockIdx.x * blockDim.x + threadIdx.x;   // (n*8 + h)
    if (i >= N * 8) return;
    int n = i >> 3, h = i & 7;
    const float4* f = (const float4*)&feat[(size_t)n * 128 + h * 16];
    const float4* a = (const float4*)&al[h * 16];
    const float4* b = (const float4*)&ar[h * 16];
    float sl = 0.f, sr = 0.f;
#pragma unroll
    for (int j = 0; j < 4; ++j) {
        float4 v = f[j], va = a[j], vb = b[j];
        sl += v.x * va.x + v.y * va.y + v.z * va.z + v.w * va.w;
        sr += v.x * vb.x + v.y * vb.y + v.z * vb.z + v.w * vb.w;
    }
    el[i] = sl; er[i] = sr;
}

// ---------------------------------------------------------------------------
// Edge softmax pass 1: segment max over dst via ordered-uint atomicMax
// ---------------------------------------------------------------------------
__global__ void gat_edge_max(const int* __restrict__ src, const int* __restrict__ dst,
                             const float* __restrict__ el, const float* __restrict__ er,
                             unsigned* __restrict__ emaxK, int E) {
    int i = blockIdx.x * blockDim.x + threadIdx.x;
    if (i >= E) return;
    int s = src[i], d = dst[i];
#pragma unroll
    for (int h = 0; h < 8; ++h) {
        float e = el[s * 8 + h] + er[d * 8 + h];
        e = (e >= 0.f) ? e : NEG_SLOPE * e;
        atomicMax(&emaxK[d * 8 + h], f2ord(e));
    }
}

// Pass 2: ex = exp(e - emax[dst]); denom[dst] += ex
__global__ void gat_edge_exp(const int* __restrict__ src, const int* __restrict__ dst,
                             const float* __restrict__ el, const float* __restrict__ er,
                             const unsigned* __restrict__ emaxK,
                             float* __restrict__ denom, float* __restrict__ ex, int E) {
    int i = blockIdx.x * blockDim.x + threadIdx.x;
    if (i >= E) return;
    int s = src[i], d = dst[i];
#pragma unroll
    for (int h = 0; h < 8; ++h) {
        float e = el[s * 8 + h] + er[d * 8 + h];
        e = (e >= 0.f) ? e : NEG_SLOPE * e;
        float m = ord2f(emaxK[d * 8 + h]);
        float v = __expf(e - m);
        ex[(size_t)i * 8 + h] = v;
        atomicAdd(&denom[d * 8 + h], v);
    }
}

// Pass 3: rst[dst,h,:] += (ex/denom[dst,h]) * feat[src,h,:]
__global__ void gat_edge_aggr(const int* __restrict__ src, const int* __restrict__ dst,
                              const float* __restrict__ ex, const float* __restrict__ denom,
                              const float* __restrict__ feat, float* __restrict__ rst, int E) {
    int i = blockIdx.x * blockDim.x + threadIdx.x;   // (edge*8 + h)
    if (i >= E * 8) return;
    int e = i >> 3, h = i & 7;
    int s = src[e], d = dst[e];
    float alpha = ex[(size_t)e * 8 + h] / denom[d * 8 + h];
    const float4* fs = (const float4*)&feat[(size_t)s * 128 + h * 16];
    float* rd = &rst[(size_t)d * 128 + h * 16];
#pragma unroll
    for (int j = 0; j < 4; ++j) {
        float4 v = fs[j];
        atomicAdd(&rd[j * 4 + 0], alpha * v.x);
        atomicAdd(&rd[j * 4 + 1], alpha * v.y);
        atomicAdd(&rd[j * 4 + 2], alpha * v.z);
        atomicAdd(&rd[j * 4 + 3], alpha * v.w);
    }
}

// h = relu(rst + bias)  (in place)
__global__ void gat_bias_relu(float* __restrict__ rst, const float* __restrict__ bias, int total) {
    int i = blockIdx.x * blockDim.x + threadIdx.x;
    if (i >= total) return;
    float v = rst[i] + bias[i & 127];
    rst[i] = v > 0.f ? v : 0.f;
}

// Column-wise partial sums over nodes -> pooled[128]
__global__ void gat_pool(const float* __restrict__ rst, float* __restrict__ pooled, int N) {
    int c = threadIdx.x;                 // 0..127
    float acc = 0.f;
    for (int r = blockIdx.x; r < N; r += gridDim.x)
        acc += rst[(size_t)r * 128 + c];
    atomicAdd(&pooled[c], acc);
}

// hg = pooled/N + b2 ; out = hg @ Wc + bc   (single block)
__global__ void gat_classify(const float* __restrict__ pooled, const float* __restrict__ b2,
                             const float* __restrict__ Wc, const float* __restrict__ bc,
                             float* __restrict__ out, int N) {
    __shared__ float hg[128];
    int c = threadIdx.x;
    hg[c] = pooled[c] / (float)N + b2[c];
    __syncthreads();
    if (c < 10) {
        float a = bc[c];
#pragma unroll 8
        for (int k = 0; k < 128; ++k) a += hg[k] * Wc[k * 10 + c];
        out[c] = a;
    }
}

// ---------------------------------------------------------------------------
// Host launch
// ---------------------------------------------------------------------------
extern "C" void kernel_launch(void* const* d_in, const int* in_sizes, int n_in,
                              void* d_out, int out_size, void* d_ws, size_t ws_size,
                              hipStream_t stream) {
    const float* x   = (const float*)d_in[0];
    const int*   src = (const int*)d_in[1];
    const int*   dst = (const int*)d_in[2];
    const float* W1  = (const float*)d_in[3];
    const float* al1 = (const float*)d_in[4];
    const float* ar1 = (const float*)d_in[5];
    const float* b1  = (const float*)d_in[6];
    const float* W2  = (const float*)d_in[7];
    const float* al2 = (const float*)d_in[8];
    const float* ar2 = (const float*)d_in[9];
    const float* b2  = (const float*)d_in[10];
    const float* Wc  = (const float*)d_in[11];
    const float* bc  = (const float*)d_in[12];
    float* out = (float*)d_out;

    const int N = in_sizes[0] / 128;
    const int E = in_sizes[1];

    char* p = (char*)d_ws;
    float*    feat   = (float*)p;    p += (size_t)N * 128 * sizeof(float);
    float*    rst    = (float*)p;    p += (size_t)N * 128 * sizeof(float);
    float*    ex     = (float*)p;    p += (size_t)E * 8   * sizeof(float);
    float*    el     = (float*)p;    p += (size_t)N * 8   * sizeof(float);
    float*    er     = (float*)p;    p += (size_t)N * 8   * sizeof(float);
    unsigned* emaxK  = (unsigned*)p; p += (size_t)N * 8   * sizeof(unsigned);
    float*    denom  = (float*)p;    p += (size_t)N * 8   * sizeof(float);
    float*    pooled = (float*)p;    p += 128 * sizeof(float);

    const dim3 gemmGrid((N + 63) / 64), gemmBlk(128);
    const int attnGrid = (N * 8 + 255) / 256;
    const int edgeGrid = (E + 255) / 256;
    const int aggrGrid = (int)(((size_t)E * 8 + 255) / 256);
    const int elemGrid = (int)(((size_t)N * 128 + 255) / 256);

    // -------- layer 1 --------
    gat_gemm_wmma<<<gemmGrid, gemmBlk, 0, stream>>>(x, W1, feat, N);
    gat_attn<<<attnGrid, 256, 0, stream>>>(feat, al1, ar1, el, er, N);
    hipMemsetAsync(emaxK, 0, (size_t)N * 8 * 4, stream);
    hipMemsetAsync(denom, 0, (size_t)N * 8 * 4, stream);
    hipMemsetAsync(rst,   0, (size_t)N * 128 * 4, stream);
    gat_edge_max <<<edgeGrid, 256, 0, stream>>>(src, dst, el, er, emaxK, E);
    gat_edge_exp <<<edgeGrid, 256, 0, stream>>>(src, dst, el, er, emaxK, denom, ex, E);
    gat_edge_aggr<<<aggrGrid, 256, 0, stream>>>(src, dst, ex, denom, feat, rst, E);
    gat_bias_relu<<<elemGrid, 256, 0, stream>>>(rst, b1, N * 128);

    // -------- layer 2 (GEMM consumes rst=h1 BEFORE rst is re-zeroed) --------
    gat_gemm_wmma<<<gemmGrid, gemmBlk, 0, stream>>>(rst, W2, feat, N);
    gat_attn<<<attnGrid, 256, 0, stream>>>(feat, al2, ar2, el, er, N);
    hipMemsetAsync(emaxK, 0, (size_t)N * 8 * 4, stream);
    hipMemsetAsync(denom, 0, (size_t)N * 8 * 4, stream);
    hipMemsetAsync(rst,   0, (size_t)N * 128 * 4, stream);
    gat_edge_max <<<edgeGrid, 256, 0, stream>>>(src, dst, el, er, emaxK, E);
    gat_edge_exp <<<edgeGrid, 256, 0, stream>>>(src, dst, el, er, emaxK, denom, ex, E);
    gat_edge_aggr<<<aggrGrid, 256, 0, stream>>>(src, dst, ex, denom, feat, rst, E);

    // -------- mean pool (+b2 folded) and classifier --------
    hipMemsetAsync(pooled, 0, 128 * 4, stream);
    gat_pool<<<256, 128, 0, stream>>>(rst, pooled, N);
    gat_classify<<<1, 128, 0, stream>>>(pooled, b2, Wc, bc, out, N);
}
